// PointNet2_34170759807380
// MI455X (gfx1250) — matrix operand order repaired and checked by reading, not compiled, not run on hardware
//
#include <hip/hip_runtime.h>
#include <hip/hip_bf16.h>
#include <cstdint>

#define BB 2              // batch
#define MAXK 768          // largest padded K (fp4: 256+512)
#define MAXC 512          // largest channel count

typedef __attribute__((ext_vector_type(16))) _Float16 v16h;
typedef __attribute__((ext_vector_type(8)))  float    v8f;

// ---------------------------------------------------------------------------
// Farthest point sampling: one block per batch, dist kept hot in L2.
// ---------------------------------------------------------------------------
__global__ void pn2_fps(const float* __restrict__ xyz, float* __restrict__ dist,
                        int* __restrict__ outIdx, int N, int npoint) {
    int b = blockIdx.x;
    xyz    += (size_t)b * N * 3;
    dist   += (size_t)b * N;
    outIdx += (size_t)b * npoint;
    int tid = threadIdx.x;
    __shared__ float sv[256];
    __shared__ int   si[256];
    __shared__ float sc[3];
    __shared__ int   sFar;
    for (int i = tid; i < N; i += blockDim.x) dist[i] = 1e10f;
    int far = 0;
    for (int it = 0; it < npoint; ++it) {
        if (tid == 0) {
            outIdx[it] = far;
            sc[0] = xyz[far * 3 + 0];
            sc[1] = xyz[far * 3 + 1];
            sc[2] = xyz[far * 3 + 2];
        }
        __syncthreads();
        float cx = sc[0], cy = sc[1], cz = sc[2];
        float bv = -1.0f; int bi = 0;
        for (int i = tid; i < N; i += blockDim.x) {
            float dx = xyz[i * 3 + 0] - cx;
            float dy = xyz[i * 3 + 1] - cy;
            float dz = xyz[i * 3 + 2] - cz;
            float d  = dx * dx + dy * dy + dz * dz;
            d = fminf(dist[i], d);
            dist[i] = d;
            if (d > bv) { bv = d; bi = i; }
        }
        sv[tid] = bv; si[tid] = bi;
        __syncthreads();
        for (int off = 128; off > 0; off >>= 1) {
            if (tid < off) {
                if (sv[tid + off] > sv[tid] ||
                    (sv[tid + off] == sv[tid] && si[tid + off] < si[tid])) {
                    sv[tid] = sv[tid + off]; si[tid] = si[tid + off];
                }
            }
            __syncthreads();
        }
        if (tid == 0) sFar = si[0];
        __syncthreads();
        far = sFar;
    }
}

// ---------------------------------------------------------------------------
// Row gather (used for new_xyz).
// ---------------------------------------------------------------------------
__global__ void pn2_gather_rows(const float* __restrict__ src, const int* __restrict__ idx,
                                float* __restrict__ dst, int S, int C, int Nin) {
    int total = BB * S * C;
    for (int i = blockIdx.x * blockDim.x + threadIdx.x; i < total;
         i += gridDim.x * blockDim.x) {
        int c = i % C;
        int s = (i / C) % S;
        int b = i / (C * S);
        int j = idx[b * S + s];
        dst[i] = src[((size_t)b * Nin + j) * C + c];
    }
}

// ---------------------------------------------------------------------------
// Ball query: one wave32 per query point; ballot/popc stream compaction,
// ascending index order (matches reference sort semantics).
// ---------------------------------------------------------------------------
__global__ void pn2_ball_query(const float* __restrict__ xyz, const float* __restrict__ nxyz,
                               int* __restrict__ idx, int N, int S, float r2, int ns) {
    int gt   = blockIdx.x * blockDim.x + threadIdx.x;
    int q    = gt >> 5;
    int lane = gt & 31;
    if (q >= BB * S) return;
    int b = q / S;
    const float* xb = xyz + (size_t)b * N * 3;
    float cx = nxyz[q * 3 + 0], cy = nxyz[q * 3 + 1], cz = nxyz[q * 3 + 2];
    int* out = idx + (size_t)q * ns;
    int cnt = 0, firstIdx = 0;
    bool haveFirst = false;
    for (int base = 0; base < N && cnt < ns; base += 32) {
        int i = base + lane;
        bool pred = false;
        if (i < N) {
            float dx = xb[i * 3 + 0] - cx;
            float dy = xb[i * 3 + 1] - cy;
            float dz = xb[i * 3 + 2] - cz;
            pred = (dx * dx + dy * dy + dz * dz) <= r2;
        }
        unsigned int mask = (unsigned int)__ballot(pred);
        if (!haveFirst && mask) { firstIdx = base + __ffs((int)mask) - 1; haveFirst = true; }
        int before = __popc(mask & ((1u << lane) - 1u));
        if (pred && (cnt + before) < ns) out[cnt + before] = i;
        cnt += __popc(mask);
    }
    if (cnt < ns)
        for (int k = cnt + lane; k < ns; k += 32) out[k] = firstIdx;
}

// ---------------------------------------------------------------------------
// Build grouped A-matrix (f16, K-padded): [xyz-centroid | point feats | 0-pad]
// ---------------------------------------------------------------------------
__global__ void pn2_build_group(const float* __restrict__ xyz, const float* __restrict__ nxyz,
                                const float* __restrict__ pts, const int* __restrict__ gidx,
                                _Float16* __restrict__ A, int S, int Nin, int ns, int Cp, int Kp) {
    int rows = BB * S * ns;
    for (int r = blockIdx.x * blockDim.x + threadIdx.x; r < rows;
         r += gridDim.x * blockDim.x) {
        int k = r % ns;
        int s = (r / ns) % S;
        int b = r / (ns * S);
        int j = gidx[((size_t)b * S + s) * ns + k];
        const float* p = xyz  + ((size_t)b * Nin + j) * 3;
        const float* c = nxyz + ((size_t)b * S   + s) * 3;
        _Float16* arow = A + (size_t)r * Kp;
        arow[0] = (_Float16)(p[0] - c[0]);
        arow[1] = (_Float16)(p[1] - c[1]);
        arow[2] = (_Float16)(p[2] - c[2]);
        int col = 3;
        if (pts) {
            const float* qf = pts + ((size_t)b * Nin + j) * Cp;
            for (int t = 0; t < Cp; ++t) arow[col++] = (_Float16)qf[t];
        }
        for (; col < Kp; ++col) arow[col] = (_Float16)0.0f;
    }
}

// ---------------------------------------------------------------------------
// Concat two f32 row-feature tensors into f16 K-padded A matrix.
// ---------------------------------------------------------------------------
__global__ void pn2_concat2(const float* __restrict__ x1, int C1,
                            const float* __restrict__ x2, int C2,
                            _Float16* __restrict__ A, int rows, int Kp) {
    size_t total = (size_t)rows * Kp;
    for (size_t i = blockIdx.x * (size_t)blockDim.x + threadIdx.x; i < total;
         i += (size_t)gridDim.x * blockDim.x) {
        int    c = (int)(i % Kp);
        size_t r = i / Kp;
        float v = 0.f;
        if (c < C1)            v = x1[r * (size_t)C1 + c];
        else if (c < C1 + C2)  v = x2[r * (size_t)C2 + (c - C1)];
        A[i] = (_Float16)v;
    }
}

// ---------------------------------------------------------------------------
// Weight convert f32 (K x N) -> f16 in this file's B-fragment order (contiguous k per half-wave, same as A):
//   frag[((t*kChunks + c)*32 + lane)*16 + i]  <-  W[k*N + n], zero K-pad
//   with k = c*32 + (lane>>4)*16 + i , n = t*16 + (lane&15)
// Each GEMM lane then reads its 16xf16 fragment as one contiguous 32B vector.
// ---------------------------------------------------------------------------
__global__ void pn2_convert_w_frag(const float* __restrict__ W, _Float16* __restrict__ Wp,
                                   int K, int N, int Kp) {
    int kChunks = Kp >> 5;
    int total   = Kp * N;                 // == N/16 * kChunks * 32 * 16
    for (int i = blockIdx.x * blockDim.x + threadIdx.x; i < total;
         i += gridDim.x * blockDim.x) {
        int e    = i & 15;
        int lane = (i >> 4) & 31;
        int rest = i >> 9;
        int c    = rest % kChunks;
        int t    = rest / kChunks;
        int k    = c * 32 + ((lane >> 4) * 16) + e;
        int n    = t * 16 + (lane & 15);
        Wp[i] = (k < K) ? (_Float16)W[(size_t)k * N + n] : (_Float16)0.0f;
    }
}

__global__ void pn2_zero_f32(float* __restrict__ p, int n) {
    for (int i = blockIdx.x * blockDim.x + threadIdx.x; i < n; i += gridDim.x * blockDim.x)
        p[i] = 0.f;
}

// ---------------------------------------------------------------------------
// WMMA GEMM: Y(MxN) = A(MxK,f16) * W(frag-order f16) + bias, fp32 accumulate.
// Block = 256 threads = 8 wave32; one 16-row A strip in LDS shared by waves,
// each wave owns one 16x16 C tile; K stepped 64 (2x v_wmma_f32_16x16x32_f16)
// with a 32-wide tail. B fragments are contiguous 32B vector loads per lane.
// ---------------------------------------------------------------------------
__global__ void __launch_bounds__(256)
pn2_wmma_gemm(const _Float16* __restrict__ A, const _Float16* __restrict__ Wfrag,
              const float* __restrict__ bias, float* __restrict__ Y,
              int M, int K, int N) {
    __shared__ _Float16 sA[16 * MAXK];
    int tid = threadIdx.x;
    int m0  = blockIdx.x * 16;
    {   // cooperative 128-bit staged copy of the contiguous 16xK strip
        const float4* Ag = reinterpret_cast<const float4*>(A + (size_t)m0 * K);
        float4*       As = reinterpret_cast<float4*>(sA);
        int n8 = (16 * K) >> 3;
        for (int i = tid; i < n8; i += 256) As[i] = Ag[i];
    }
    __syncthreads();
    int wave = tid >> 5;
    int lane = tid & 31;
    int tn   = blockIdx.y * 8 + wave;
    if (tn * 16 >= N) return;           // wave-uniform: EXEC stays all-ones below

    int m       = lane & 15;            // A tile row for this lane
    int kh      = (lane >> 4) * 16;     // A: upper half-wave holds K+16 group
    int kChunks = K >> 5;
    // per-lane fragment stream for this N tile: kChunks fragments of 16 halfs
    const v16h* wfr = reinterpret_cast<const v16h*>(
        Wfrag + ((size_t)tn * kChunks * 32 + lane) * 16);

    v8f acc = {};
    int cidx = 0;
#define PN2_AFRAG(af_, k0_)                                                    \
    {                                                                          \
        _Pragma("unroll")                                                      \
        for (int j = 0; j < 8; ++j) {                                          \
            int kk = (k0_) + kh + 2 * j;                                       \
            (af_)[2 * j]     = sA[m * K + kk];                                 \
            (af_)[2 * j + 1] = sA[m * K + kk + 1];                            \
        }                                                                      \
    }
    for (; cidx + 2 <= kChunks; cidx += 2) {
        v16h af0, af1;
        PN2_AFRAG(af0, cidx * 32)
        PN2_AFRAG(af1, cidx * 32 + 32)
        v16h bf0 = wfr[(size_t)cidx * 32];        // stride between chunks: 32 frags
        v16h bf1 = wfr[(size_t)(cidx + 1) * 32];
        if (cidx + 2 < kChunks)                    // gfx1250 global_prefetch_b8
            __builtin_prefetch((const void*)&wfr[(size_t)(cidx + 2) * 32], 0, 1);
        acc = __builtin_amdgcn_wmma_f32_16x16x32_f16(false, af0, false, bf0,
                                                     (short)0, acc, false, false);
        acc = __builtin_amdgcn_wmma_f32_16x16x32_f16(false, af1, false, bf1,
                                                     (short)0, acc, false, false);
    }
    if (cidx < kChunks) {
        v16h af0;
        PN2_AFRAG(af0, cidx * 32)
        v16h bf0 = wfr[(size_t)cidx * 32];
        acc = __builtin_amdgcn_wmma_f32_16x16x32_f16(false, af0, false, bf0,
                                                     (short)0, acc, false, false);
    }
#undef PN2_AFRAG

    int   ncol = tn * 16 + (lane & 15);
    float bb   = bias[ncol];
    int   mrow = m0 + (lane >> 4) * 8;  // C layout: VGPR r -> M=r / r+8
#pragma unroll
    for (int r = 0; r < 8; ++r)
        Y[(size_t)(mrow + r) * N + ncol] = acc[r] + bb;
}

// ---------------------------------------------------------------------------
// BatchNorm statistics: per-channel sum/sumsq (LDS partials, one flush/block)
// ---------------------------------------------------------------------------
__global__ void pn2_bn_stats(const float* __restrict__ Y, int M, int N,
                             float* __restrict__ stats) {
    __shared__ float ls[2 * MAXC];
    int tid = threadIdx.x;
    for (int i = tid; i < 2 * N; i += blockDim.x) ls[i] = 0.f;
    __syncthreads();
    size_t total = (size_t)M * N;
    for (size_t i = blockIdx.x * (size_t)blockDim.x + tid; i < total;
         i += (size_t)gridDim.x * blockDim.x) {
        float v = Y[i];
        int   c = (int)(i % N);
        atomicAdd(&ls[2 * c],     v);
        atomicAdd(&ls[2 * c + 1], v * v);
    }
    __syncthreads();
    for (int i = tid; i < 2 * N; i += blockDim.x) atomicAdd(&stats[i], ls[i]);
}

// BN + ReLU -> f16 K-padded next-layer A matrix
__global__ void pn2_bn_relu16(const float* __restrict__ Y, const float* __restrict__ stats,
                              const float* __restrict__ gamma, const float* __restrict__ beta,
                              int M, int N, int Kp, float invM, _Float16* __restrict__ out) {
    size_t total = (size_t)M * Kp;
    for (size_t i = blockIdx.x * (size_t)blockDim.x + threadIdx.x; i < total;
         i += (size_t)gridDim.x * blockDim.x) {
        int    c = (int)(i % Kp);
        size_t r = i / Kp;
        float v = 0.f;
        if (c < N) {
            float x    = Y[r * (size_t)N + c];
            float mean = stats[2 * c] * invM;
            float var  = stats[2 * c + 1] * invM - mean * mean;
            v = (x - mean) * rsqrtf(var + 1e-5f) * gamma[c] + beta[c];
            v = fmaxf(v, 0.f);
        }
        out[i] = (_Float16)v;
    }
}

// BN + ReLU -> f32 (final layer of an MLP); in-place safe (elementwise)
__global__ void pn2_bn_relu32(const float* __restrict__ Y, const float* __restrict__ stats,
                              const float* __restrict__ gamma, const float* __restrict__ beta,
                              int M, int N, float invM, float* __restrict__ out) {
    size_t total = (size_t)M * N;
    for (size_t i = blockIdx.x * (size_t)blockDim.x + threadIdx.x; i < total;
         i += (size_t)gridDim.x * blockDim.x) {
        int   c    = (int)(i % N);
        float x    = Y[i];
        float mean = stats[2 * c] * invM;
        float var  = stats[2 * c + 1] * invM - mean * mean;
        float v    = (x - mean) * rsqrtf(var + 1e-5f) * gamma[c] + beta[c];
        out[i] = fmaxf(v, 0.f);
    }
}

// Max over nsample dimension: (rows, ns, C) -> (rows, C)
__global__ void pn2_maxpool(const float* __restrict__ x, float* __restrict__ out,
                            int rows, int ns, int C) {
    int total = rows * C;
    for (int i = blockIdx.x * blockDim.x + threadIdx.x; i < total;
         i += gridDim.x * blockDim.x) {
        int c = i % C, r = i / C;
        const float* p = x + ((size_t)r * ns) * C + c;
        float m = p[0];
        for (int k = 1; k < ns; ++k) m = fmaxf(m, p[(size_t)k * C]);
        out[i] = m;
    }
}

// ---------------------------------------------------------------------------
// 3-NN inverse-distance interpolation (feature propagation upsample)
// ---------------------------------------------------------------------------
__global__ void pn2_three_interp(const float* __restrict__ x1, const float* __restrict__ x2,
                                 const float* __restrict__ p2, float* __restrict__ out,
                                 int Nn, int S, int C) {
    int total = BB * Nn;
    for (int i = blockIdx.x * blockDim.x + threadIdx.x; i < total;
         i += gridDim.x * blockDim.x) {
        int b = i / Nn;
        float px = x1[(size_t)i * 3 + 0];
        float py = x1[(size_t)i * 3 + 1];
        float pz = x1[(size_t)i * 3 + 2];
        const float* xb = x2 + (size_t)b * S * 3;
        float d0 = 1e30f, d1 = 1e30f, d2 = 1e30f;
        int   i0 = 0, i1 = 0, i2 = 0;
        for (int j = 0; j < S; ++j) {
            float dx = xb[j * 3 + 0] - px;
            float dy = xb[j * 3 + 1] - py;
            float dz = xb[j * 3 + 2] - pz;
            float d  = dx * dx + dy * dy + dz * dz;
            if (d < d0)      { d2 = d1; i2 = i1; d1 = d0; i1 = i0; d0 = d; i0 = j; }
            else if (d < d1) { d2 = d1; i2 = i1; d1 = d;  i1 = j; }
            else if (d < d2) { d2 = d;  i2 = j; }
        }
        float w0 = 1.f / (d0 + 1e-8f), w1 = 1.f / (d1 + 1e-8f), w2 = 1.f / (d2 + 1e-8f);
        float ws = w0 + w1 + w2; w0 /= ws; w1 /= ws; w2 /= ws;
        const float* r0 = p2 + ((size_t)b * S + i0) * C;
        const float* r1 = p2 + ((size_t)b * S + i1) * C;
        const float* r2 = p2 + ((size_t)b * S + i2) * C;
        float* o = out + (size_t)i * C;
        for (int c = 0; c < C; ++c) o[c] = w0 * r0[c] + w1 * r1[c] + w2 * r2[c];
    }
}

// Final writeout: [global feats (B,512)] ++ [transpose(lfin) -> (B,64,N)]
__global__ void pn2_write_out(const float* __restrict__ g, const float* __restrict__ pf,
                              float* __restrict__ out, int N, int C) {
    size_t gsz   = (size_t)BB * 512;
    size_t total = gsz + (size_t)BB * C * N;
    for (size_t i = blockIdx.x * (size_t)blockDim.x + threadIdx.x; i < total;
         i += (size_t)gridDim.x * blockDim.x) {
        if (i < gsz) { out[i] = g[i]; continue; }
        size_t j = i - gsz;
        int n = (int)(j % N);
        int c = (int)((j / N) % C);
        int b = (int)(j / ((size_t)N * C));
        out[i] = pf[((size_t)b * N + n) * C + c];
    }
}

// ===========================================================================
extern "C" void kernel_launch(void* const* d_in, const int* in_sizes, int n_in,
                              void* d_out, int out_size, void* d_ws, size_t ws_size,
                              hipStream_t stream) {
    (void)in_sizes; (void)n_in; (void)out_size; (void)ws_size;
    const float* xyz0 = (const float*)d_in[0];

    // ---- workspace carving ------------------------------------------------
    char* cur = (char*)d_ws;
    auto alloc = [&](size_t bytes) -> void* {
        void* p = (void*)cur;
        cur += (bytes + 255) & ~(size_t)255;
        return p;
    };
    float* dist    = (float*)alloc((size_t)BB * 16384 * 4);
    int*   fpsIdx  = (int*)  alloc((size_t)BB * 4096 * 4);
    int*   ballIdx = (int*)  alloc((size_t)BB * 4096 * 32 * 4);
    float* l0x  = (float*)alloc((size_t)BB * 4096 * 3 * 4);
    float* l05x = (float*)alloc((size_t)BB * 2048 * 3 * 4);
    float* l1x  = (float*)alloc((size_t)BB * 1024 * 3 * 4);
    float* l2x  = (float*)alloc((size_t)BB * 256  * 3 * 4);
    float* l3x  = (float*)alloc((size_t)BB * 64   * 3 * 4);
    float* l4x  = (float*)alloc((size_t)BB * 16   * 3 * 4);
    float* l0p  = (float*)alloc((size_t)BB * 4096 * 64  * 4);
    float* l05p = (float*)alloc((size_t)BB * 2048 * 128 * 4);
    float* l1p  = (float*)alloc((size_t)BB * 1024 * 128 * 4);
    float* l2p  = (float*)alloc((size_t)BB * 256  * 256 * 4);
    float* l3p  = (float*)alloc((size_t)BB * 64   * 256 * 4);
    float* l4p  = (float*)alloc((size_t)BB * 16   * 512 * 4);
    float* l5p  = (float*)alloc((size_t)BB * 512 * 4);
    float* lfin = (float*)alloc((size_t)BB * 16384 * 64 * 4);
    _Float16* Abuf  = (_Float16*)alloc((size_t)8388608 * 2);
    _Float16* A2buf = (_Float16*)alloc((size_t)8388608 * 2);
    _Float16* Wbuf  = (_Float16*)alloc((size_t)MAXK * MAXC * 2);
    float* Ybuf   = (float*)alloc((size_t)8388608 * 4);
    float* stats  = (float*)alloc((size_t)2 * MAXC * 4);
    float* interp = (float*)alloc((size_t)BB * 16384 * 64 * 4);

    auto gs   = [](size_t n, int b) { return (unsigned)((n + (size_t)b - 1) / b); };
    auto kpad = [](int k) { return (k + 31) & ~31; };
    int pc = 1;   // param cursor into d_in (W,b,gamma,beta per layer)

    // ---- generic WMMA MLP runner -----------------------------------------
    auto run_mlp = [&](_Float16* Ain, int M, int cin0, const int* widths, int nl,
                       float* finalOut) {
        _Float16* Acur = Ain;
        _Float16* Aalt = (Ain == Abuf) ? A2buf : Abuf;
        int cin = cin0;
        for (int li = 0; li < nl; ++li) {
            int Kp = kpad(cin);
            int Nc = widths[li];
            const float* W  = (const float*)d_in[pc++];
            const float* bb = (const float*)d_in[pc++];
            const float* gg = (const float*)d_in[pc++];
            const float* be = (const float*)d_in[pc++];
            pn2_convert_w_frag<<<gs((size_t)Kp * Nc, 256), 256, 0, stream>>>(
                W, Wbuf, cin, Nc, Kp);
            dim3 grid(M / 16, (Nc + 127) / 128);
            pn2_wmma_gemm<<<grid, 256, 0, stream>>>(Acur, Wbuf, bb, Ybuf, M, Kp, Nc);
            pn2_zero_f32<<<gs(2 * Nc, 256), 256, 0, stream>>>(stats, 2 * Nc);
            pn2_bn_stats<<<256, 256, 0, stream>>>(Ybuf, M, Nc, stats);
            float invM = 1.0f / (float)M;
            if (li + 1 < nl) {
                int Kpn = kpad(Nc);
                pn2_bn_relu16<<<gs((size_t)M * Kpn, 256), 256, 0, stream>>>(
                    Ybuf, stats, gg, be, M, Nc, Kpn, invM, Aalt);
                _Float16* t = Acur; Acur = Aalt; Aalt = t;
            } else {
                pn2_bn_relu32<<<gs((size_t)M * Nc, 256), 256, 0, stream>>>(
                    Ybuf, stats, gg, be, M, Nc, invM, finalOut);
            }
            cin = Nc;
        }
    };

    // ---- set abstraction level -------------------------------------------
    auto sa_level = [&](const float* xin, const float* pin, int Nin, int np, float rad,
                        int Cp, const int* widths, int nl, float* xout, float* pout) {
        pn2_fps<<<BB, 256, 0, stream>>>(xin, dist, fpsIdx, Nin, np);
        pn2_gather_rows<<<gs((size_t)BB * np * 3, 256), 256, 0, stream>>>(
            xin, fpsIdx, xout, np, 3, Nin);
        pn2_ball_query<<<gs((size_t)BB * np * 32, 256), 256, 0, stream>>>(
            xin, xout, ballIdx, Nin, np, rad * rad, 32);
        int M  = BB * np * 32;
        int Kp = kpad(3 + Cp);
        pn2_build_group<<<gs((size_t)M, 256), 256, 0, stream>>>(
            xin, xout, pin, ballIdx, Abuf, np, Nin, 32, Cp, Kp);
        run_mlp(Abuf, M, 3 + Cp, widths, nl, Ybuf);
        int lastC = widths[nl - 1];
        pn2_maxpool<<<gs((size_t)BB * np * lastC, 256), 256, 0, stream>>>(
            Ybuf, pout, BB * np, 32, lastC);
    };

    // ---- feature propagation level ---------------------------------------
    auto fp_level = [&](const float* x1, const float* x2, const float* p1, int C1,
                        const float* p2, int C2, int Nn, int S,
                        const int* widths, int nl, float* pout) {
        pn2_three_interp<<<gs((size_t)BB * Nn, 256), 256, 0, stream>>>(
            x1, x2, p2, interp, Nn, S, C2);
        int M  = BB * Nn;
        int Kp = kpad(C1 + C2);
        pn2_concat2<<<gs((size_t)M * Kp, 256), 256, 0, stream>>>(
            p1, C1, interp, C2, Abuf, M, Kp);
        run_mlp(Abuf, M, C1 + C2, widths, nl, pout);
    };

    // ---- network ----------------------------------------------------------
    const int w_sa0[3]  = {16, 16, 32};
    const int w_sa05[3] = {32, 32, 32};
    const int w_sa1[3]  = {32, 32, 64};
    const int w_sa2[3]  = {64, 64, 128};
    const int w_sa3[3]  = {128, 128, 256};
    const int w_sa4[3]  = {256, 256, 512};
    const int w_sa5[3]  = {256, 256, 512};
    const int w_fp4[2]  = {256, 256};
    const int w_fp3[2]  = {256, 256};
    const int w_fp2[2]  = {256, 128};
    const int w_fp1[3]  = {128, 128, 128};
    const int w_fp05[3] = {128, 128, 64};
    const int w_fp0[3]  = {128, 128, 64};

    sa_level(xyz0, nullptr, 16384, 4096, 0.1f, 0,   w_sa0,  3, l0x,  l0p);
    sa_level(l0x,  l0p,     4096,  2048, 0.1f, 32,  w_sa05, 3, l05x, l05p);
    sa_level(l05x, l05p,    2048,  1024, 0.1f, 32,  w_sa1,  3, l1x,  l1p);
    sa_level(l1x,  l1p,     1024,  256,  0.2f, 64,  w_sa2,  3, l2x,  l2p);
    sa_level(l2x,  l2p,     256,   64,   0.4f, 128, w_sa3,  3, l3x,  l3p);
    sa_level(l3x,  l3p,     64,    16,   0.8f, 256, w_sa4,  3, l4x,  l4p);

    {   // set_abstraction_all (sa5) over the 64 l3 points
        int M = BB * 64, Kp = kpad(259);
        pn2_concat2<<<gs((size_t)M * Kp, 256), 256, 0, stream>>>(
            l3x, 3, l3p, 256, Abuf, M, Kp);
        run_mlp(Abuf, M, 259, w_sa5, 3, Ybuf);
        pn2_maxpool<<<gs((size_t)BB * 512, 256), 256, 0, stream>>>(Ybuf, l5p, BB, 64, 512);
    }

    fp_level(l3x,  l4x,  l3p,  256, l4p,  512, 64,    16,   w_fp4,  2, l3p);
    fp_level(l2x,  l3x,  l2p,  128, l3p,  256, 256,   64,   w_fp3,  2, l2p);
    fp_level(l1x,  l2x,  l1p,  64,  l2p,  256, 1024,  256,  w_fp2,  2, l1p);
    fp_level(l05x, l1x,  l05p, 32,  l1p,  128, 2048,  1024, w_fp1,  3, l05p);
    fp_level(l0x,  l05x, l0p,  32,  l05p, 128, 4096,  2048, w_fp05, 3, l0p);
    fp_level(xyz0, l0x,  nullptr, 0, l0p, 64,  16384, 4096, w_fp0,  3, lfin);

    pn2_write_out<<<gs((size_t)BB * 512 + (size_t)BB * 64 * 16384, 256), 256, 0, stream>>>(
        l5p, lfin, (float*)d_out, 16384, 64);
}
